// PointsFusion_17755394801930
// MI455X (gfx1250) — matrix operand order repaired and checked by reading, not compiled, not run on hardware
//
#include <hip/hip_runtime.h>
#include <math.h>

// PointsFusion for MI455X (gfx1250, wave32, WMMA, TDM).
// init_weights: fold BN, pre-swizzle f16 WMMA A-fragments.
// knn_kernel:   TDM-stage candidate coords to LDS; d^2 via V_WMMA_F32_16X16X4_F32;
//               full 16x2048 distance rows in LDS; 32-round parallel argmin.
// fuse_kernel:  per-point 3-layer MLP; activations kept in B-fragment order so
//               every WMMA B operand is 2x ds_load_b128 and every epilogue is
//               1x ds_store_b128; fused channel-max + softmax + weighted sum.

#define BN_EPS 1e-3f
#define NPTS   2048
#define KNN    32
#define QT     16

typedef __attribute__((ext_vector_type(2)))  float        v2f;
typedef __attribute__((ext_vector_type(8)))  float        v8f;
typedef __attribute__((ext_vector_type(16))) _Float16     v16h;
typedef __attribute__((ext_vector_type(4)))  unsigned int u32x4;
typedef __attribute__((ext_vector_type(8)))  int          i32x8;
typedef __attribute__((ext_vector_type(4)))  int          i32x4;

union V2F  { v2f  v; float    f[2];  };
union V8F  { v8f  v; float    f[8];  };
union V16H { v16h v; _Float16 h[16]; uint4 q[2]; };
union PK8  { _Float16 h[8]; uint4 q; };

// ---------------------------------------------------------------------------
// Kernel 1: fold BN into conv weights; pre-swizzle W2/W3 into WMMA A-fragment
// lane order (A 16x32 f16, contiguous K as in Xb: lanes 0-15 hold K=0..15, lanes 16-31 hold
// K=16..31; M = mo*16 + (lane&15)) so fuse loads one v16h per lane.
// ---------------------------------------------------------------------------
__global__ __launch_bounds__(256) void init_weights_kernel(
    const float* __restrict__ W1, const float* __restrict__ b1,
    const float* __restrict__ g1, const float* __restrict__ be1,
    const float* __restrict__ m1, const float* __restrict__ v1,
    const float* __restrict__ W2, const float* __restrict__ b2,
    const float* __restrict__ g2, const float* __restrict__ be2,
    const float* __restrict__ m2, const float* __restrict__ v2,
    const float* __restrict__ W3, const float* __restrict__ b3,
    const float* __restrict__ g3, const float* __restrict__ be3,
    const float* __restrict__ m3, const float* __restrict__ v3,
    float* __restrict__ W1f, float* __restrict__ bb1,
    float* __restrict__ bb2, float* __restrict__ bb3,
    _Float16* __restrict__ W2A, _Float16* __restrict__ W3A)
{
    const int tid = threadIdx.x;
    if (tid < 64) {
        const float s1 = g1[tid] * rsqrtf(v1[tid] + BN_EPS);
        for (int c = 0; c < 4; ++c) W1f[tid * 4 + c] = W1[tid * 4 + c] * s1;
        bb1[tid] = (b1[tid] - m1[tid]) * s1 + be1[tid];
        const float s2 = g2[tid] * rsqrtf(v2[tid] + BN_EPS);
        bb2[tid] = (b2[tid] - m2[tid]) * s2 + be2[tid];
    }
    if (tid < 128) {
        const float s3 = g3[tid] * rsqrtf(v3[tid] + BN_EPS);
        bb3[tid] = (b3[tid] - m3[tid]) * s3 + be3[tid];
    }
    for (int ii = tid; ii < 64 * 64; ii += blockDim.x) {
        const int h = ii & 15, ln = (ii >> 4) & 31, tIdx = ii >> 9;
        const int kt = tIdx & 1, mo = tIdx >> 1;
        const int row = mo * 16 + (ln & 15);
        const int koff = (ln >= 16) ? 16 : 0;
        const int K = kt * 32 + koff + h;
        const float s2 = g2[row] * rsqrtf(v2[row] + BN_EPS);
        W2A[ii] = (_Float16)(W2[row * 64 + K] * s2);
    }
    for (int ii = tid; ii < 128 * 64; ii += blockDim.x) {
        const int h = ii & 15, ln = (ii >> 4) & 31, tIdx = ii >> 9;
        const int kt = tIdx & 1, mo = tIdx >> 1;
        const int row = mo * 16 + (ln & 15);
        const int koff = (ln >= 16) ? 16 : 0;
        const int K = kt * 32 + koff + h;
        const float s3 = g3[row] * rsqrtf(v3[row] + BN_EPS);
        W3A[ii] = (_Float16)(W3[row * 64 + K] * s3);
    }
}

// ---------------------------------------------------------------------------
// Kernel 2: KNN. TDM stages the target's 3x2048 coords into LDS once, then
// d2 = |q|^2 - 2 q.c + |c|^2 via V_WMMA_F32_16X16X4_F32 with
//   A row m = [-2qx, -2qy, -2qz, |q|^2], B col n = [cx, cy, cz, 1],
// plus |c|^2 per lane (all 8 D regs of a lane share one column N).
// ---------------------------------------------------------------------------
__global__ __launch_bounds__(128) void knn_kernel(
    const float* __restrict__ points1, const float* __restrict__ points2,
    int* __restrict__ knn_idx /* [2][B][N][KNN] */, int B, int N)
{
    extern __shared__ float d2s[];               // [QT][N] = 128 KB dynamic
    __shared__ float cbuf[3][NPTS];              // TDM-staged candidate coords
    __shared__ float redV[QT][8];
    __shared__ int   redI[QT][8];

    const int tid  = threadIdx.x;
    const int wave = tid >> 5;
    const int lane = tid & 31;
    const int hi   = lane >> 4;
    const int lm   = lane & 15;

    const int q0 = blockIdx.x * QT;
    const int b  = blockIdx.y;
    const int t  = blockIdx.z;
    const float* tp = (t == 0) ? points1 : points2;

    // --- TDM: DMA the 2-D tile (dim0=2048 floats, dim1=3 rows) into LDS. ---
    if (wave == 0) {
        const unsigned long long ga =
            (unsigned long long)(const void*)(tp + (size_t)b * 3 * N);
        const unsigned ldsoff = (unsigned)(unsigned long long)(&cbuf[0][0]);
        const u32x4 g0 = {
            1u,                                        // count=1 (user D#)
            ldsoff,                                    // lds_addr (bytes)
            (unsigned)(ga & 0xffffffffu),              // global_addr[31:0]
            (unsigned)((ga >> 32) & 0x1ffffffu)        // global_addr[56:32]
                | 0x80000000u                          // type=2 ("image")
        };
        const i32x8 g1 = {
            (int)(2u << 16),               // data_size=2 (4 bytes)
            (int)((unsigned)NPTS << 16),   // tensor_dim0[15:0] << 16
            (int)(3u << 16),               // tensor_dim0 hi=0 | tensor_dim1=3
            (int)((unsigned)NPTS << 16),   // tile_dim0 = 2048
            3,                             // tile_dim1=3, tile_dim2=0
            NPTS,                          // tensor_dim0_stride lo32
            (int)((unsigned)NPTS << 16),   // stride hi=0 | tensor_dim1_stride lo16
            0
        };
        const i32x4 z4 = { 0, 0, 0, 0 };
        const i32x8 z8 = { 0, 0, 0, 0, 0, 0, 0, 0 };
        __builtin_amdgcn_tensor_load_to_lds(g0, g1, z4, z4, z8, 0);
        __builtin_amdgcn_s_wait_tensorcnt(0);
    }
    __syncthreads();

    // A fragment (16x4 f32): lanes 0-15 hold K=0,1; lanes 16-31 hold K=2,3.
    const int   qn  = q0 + lm;
    const float qx  = points1[(b * 3 + 0) * N + qn];
    const float qy  = points1[(b * 3 + 1) * N + qn];
    const float qz  = points1[(b * 3 + 2) * N + qn];
    const float qn2 = qx * qx + qy * qy + qz * qz;
    V2F a;
    a.f[0] = hi ? (-2.0f * qz) : (-2.0f * qx);
    a.f[1] = hi ? qn2          : (-2.0f * qy);

    for (int ct = wave; ct < (N >> 4); ct += 4) {
        const int   cand = ct * 16 + lm;
        const float cx = cbuf[0][cand];
        const float cy = cbuf[1][cand];
        const float cz = cbuf[2][cand];
        const float c2 = cx * cx + cy * cy + cz * cz;
        V2F bf;
        bf.f[0] = hi ? cz   : cx;
        bf.f[1] = hi ? 1.0f : cy;
        V8F acc = {};
        acc.v = __builtin_amdgcn_wmma_f32_16x16x4_f32(
            false, a.v, false, bf.v, (short)0, acc.v, false, false);
#pragma unroll
        for (int r = 0; r < 8; ++r)
            d2s[(hi * 8 + r) * N + cand] = acc.f[r] + c2;
    }
    __syncthreads();

    // top-32: 8 threads per query, 32 extraction rounds.
    const int qq  = tid >> 3;
    const int sub = tid & 7;
    const int seg = N >> 3;
    float* row = &d2s[qq * N];
    for (int j = 0; j < KNN; ++j) {
        float bv = 3.4e38f;
        int   bi = sub * seg;
        for (int c = sub * seg; c < sub * seg + seg; ++c) {
            const float v = row[c];
            if (v < bv) { bv = v; bi = c; }
        }
        redV[qq][sub] = bv;
        redI[qq][sub] = bi;
        __syncthreads();
        if (sub == 0) {
            float best = redV[qq][0]; int besti = redI[qq][0];
            for (int s = 1; s < 8; ++s)
                if (redV[qq][s] < best) { best = redV[qq][s]; besti = redI[qq][s]; }
            knn_idx[((size_t)(t * B + b) * N + (q0 + qq)) * KNN + j] = besti;
            row[besti] = 3.0e38f;
        }
        __syncthreads();
    }
}

// ---------------------------------------------------------------------------
// Kernel 3: per-point fused MLP + channel-max + softmax + weighted sum.
// Activations stored in B-fragment order Xb[no][kt][lane][16]:
//   element (no,kt,lane,h) == X[channel = kt*32 + (lane>>4)*16 + h]
//                              [col     = no*16 + (lane&15)]
// so a B operand is two ds_load_b128 and a D epilogue is one ds_store_b128.
// ---------------------------------------------------------------------------
__global__ __launch_bounds__(32) void fuse_kernel(
    const float* __restrict__ points1, const float* __restrict__ points2,
    const int* __restrict__ knn_idx,
    const float* __restrict__ W1f, const float* __restrict__ bb1,
    const _Float16* __restrict__ W2A, const float* __restrict__ bb2,
    const _Float16* __restrict__ W3A, const float* __restrict__ bb3,
    float* __restrict__ out, int B, int N)
{
    __shared__ float nnc[3][64];
    __shared__ float feat[4][64];
    __shared__ __align__(16) _Float16 X1b[4][2][32][16];
    __shared__ __align__(16) _Float16 X2b[4][2][32][16];
    __shared__ unsigned colmax[64];

    const int lane = threadIdx.x;
    const int n = blockIdx.x, b = blockIdx.y;
    const int hi = lane >> 4;
    const int lm = lane & 15;

    __builtin_prefetch(W2A, 0, 2);   // global_prefetch (weights L2-hot)
    __builtin_prefetch(W3A, 0, 2);

    const float qx = points1[(b * 3 + 0) * N + n];
    const float qy = points1[(b * 3 + 1) * N + n];
    const float qz = points1[(b * 3 + 2) * N + n];

    colmax[lane] = 0u;
    colmax[lane + 32] = 0u;

    // Gather 64 neighbors (32 self + 32 cross), feat = [resi, dist].
#pragma unroll
    for (int rep = 0; rep < 2; ++rep) {
        const int j  = lane + rep * 32;
        const int t  = j >> 5, jj = j & 31;
        const int idx = knn_idx[((size_t)(t * B + b) * N + n) * KNN + jj];
        const float* tp = t ? points2 : points1;
        const float cx = tp[(b * 3 + 0) * N + idx];
        const float cy = tp[(b * 3 + 1) * N + idx];
        const float cz = tp[(b * 3 + 2) * N + idx];
        const float rx = cx - qx, ry = cy - qy, rz = cz - qz;
        const float ss = rx * rx + ry * ry + rz * rz;
        nnc[0][j] = cx; nnc[1][j] = cy; nnc[2][j] = cz;
        feat[0][j] = rx; feat[1][j] = ry; feat[2][j] = rz;
        feat[3][j] = sqrtf(fmaxf(ss, 1e-12f));
    }
    __syncthreads();

    // Layer 1: 4 -> 64 scalar f32 (K=4 too small for WMMA); 8 consecutive
    // channels per thread-iteration -> one 16-byte LDS store into X1b.
    for (int ii = lane; ii < 512; ii += 32) {
        const int j  = ii >> 3;
        const int o0 = (ii & 7) * 8;
        PK8 pk;
#pragma unroll
        for (int r = 0; r < 8; ++r) {
            const int o = o0 + r;
            const float v = bb1[o]
                + W1f[o * 4 + 0] * feat[0][j] + W1f[o * 4 + 1] * feat[1][j]
                + W1f[o * 4 + 2] * feat[2][j] + W1f[o * 4 + 3] * feat[3][j];
            pk.h[r] = (_Float16)fmaxf(v, 0.0f);
        }
        const int no = j >> 4, kt = o0 >> 5;
        const int lpos = (j & 15) + 16 * ((o0 >> 4) & 1);
        *(uint4*)&X1b[no][kt][lpos][o0 & 15] = pk.q;
    }
    __syncthreads();

    // Layer 2: 64 -> 64. A fragments hoisted; no-loop unrolled for 4
    // independent accumulators per weight tile.
    for (int mo = 0; mo < 4; ++mo) {
        const v16h a0 = *(const v16h*)(W2A + ((size_t)(mo * 2 + 0) * 32 + lane) * 16);
        const v16h a1 = *(const v16h*)(W2A + ((size_t)(mo * 2 + 1) * 32 + lane) * 16);
        const int mbase = mo * 16 + hi * 8;
        const float4 bv0 = *(const float4*)(bb2 + mbase);
        const float4 bv1 = *(const float4*)(bb2 + mbase + 4);
        const float bvf[8] = { bv0.x, bv0.y, bv0.z, bv0.w, bv1.x, bv1.y, bv1.z, bv1.w };
#pragma unroll
        for (int no = 0; no < 4; ++no) {
            V16H bf0, bf1;
            bf0.q[0] = *(const uint4*)&X1b[no][0][lane][0];
            bf0.q[1] = *(const uint4*)&X1b[no][0][lane][8];
            bf1.q[0] = *(const uint4*)&X1b[no][1][lane][0];
            bf1.q[1] = *(const uint4*)&X1b[no][1][lane][8];
            V8F acc = {};
            acc.v = __builtin_amdgcn_wmma_f32_16x16x32_f16(
                false, a0, false, bf0.v, (short)0, acc.v, false, false);
            acc.v = __builtin_amdgcn_wmma_f32_16x16x32_f16(
                false, a1, false, bf1.v, (short)0, acc.v, false, false);
            PK8 pk;
#pragma unroll
            for (int r = 0; r < 8; ++r)
                pk.h[r] = (_Float16)fmaxf(acc.f[r] + bvf[r], 0.0f);
            *(uint4*)&X2b[no][mo >> 1][lm + 16 * (mo & 1)][hi * 8] = pk.q;
        }
    }
    __syncthreads();

    // Layer 3: 64 -> 128, channel-max fused (X3 never materialized).
    for (int mo = 0; mo < 8; ++mo) {
        const v16h a0 = *(const v16h*)(W3A + ((size_t)(mo * 2 + 0) * 32 + lane) * 16);
        const v16h a1 = *(const v16h*)(W3A + ((size_t)(mo * 2 + 1) * 32 + lane) * 16);
        const int mbase = mo * 16 + hi * 8;
        const float4 bv0 = *(const float4*)(bb3 + mbase);
        const float4 bv1 = *(const float4*)(bb3 + mbase + 4);
        const float bvf[8] = { bv0.x, bv0.y, bv0.z, bv0.w, bv1.x, bv1.y, bv1.z, bv1.w };
#pragma unroll
        for (int no = 0; no < 4; ++no) {
            V16H bf0, bf1;
            bf0.q[0] = *(const uint4*)&X2b[no][0][lane][0];
            bf0.q[1] = *(const uint4*)&X2b[no][0][lane][8];
            bf1.q[0] = *(const uint4*)&X2b[no][1][lane][0];
            bf1.q[1] = *(const uint4*)&X2b[no][1][lane][8];
            V8F acc = {};
            acc.v = __builtin_amdgcn_wmma_f32_16x16x32_f16(
                false, a0, false, bf0.v, (short)0, acc.v, false, false);
            acc.v = __builtin_amdgcn_wmma_f32_16x16x32_f16(
                false, a1, false, bf1.v, (short)0, acc.v, false, false);
            float mx = 0.0f;
#pragma unroll
            for (int r = 0; r < 8; ++r)
                mx = fmaxf(mx, fmaxf(acc.f[r] + bvf[r], 0.0f));
            // relu >= 0 so float bit order == uint order.
            atomicMax(&colmax[no * 16 + lm], __float_as_uint(mx));
        }
    }
    __syncthreads();

    // Softmax over 64 neighbors, weighted sum of neighbor coords.
    const float v0 = __uint_as_float(colmax[lane]);
    const float v1 = __uint_as_float(colmax[lane + 32]);
    float m = fmaxf(v0, v1);
    for (int off = 16; off > 0; off >>= 1) m = fmaxf(m, __shfl_xor(m, off));
    const float e0 = __expf(v0 - m), e1 = __expf(v1 - m);
    float s = e0 + e1;
    for (int off = 16; off > 0; off >>= 1) s += __shfl_xor(s, off);
    const float w0 = e0 / s, w1 = e1 / s;
#pragma unroll
    for (int d = 0; d < 3; ++d) {
        float p = w0 * nnc[d][lane] + w1 * nnc[d][lane + 32];
        for (int off = 16; off > 0; off >>= 1) p += __shfl_xor(p, off);
        if (lane == 0) out[(size_t)(b * 3 + d) * N + n] = p;
    }
}

// ---------------------------------------------------------------------------
extern "C" void kernel_launch(void* const* d_in, const int* in_sizes, int n_in,
                              void* d_out, int out_size, void* d_ws, size_t ws_size,
                              hipStream_t stream) {
    (void)n_in; (void)out_size; (void)ws_size;
    const float* points1 = (const float*)d_in[0];
    const float* points2 = (const float*)d_in[1];
    const float* W1 = (const float*)d_in[2];
    const float* b1 = (const float*)d_in[3];
    const float* g1 = (const float*)d_in[4];
    const float* be1 = (const float*)d_in[5];
    const float* m1 = (const float*)d_in[6];
    const float* v1 = (const float*)d_in[7];
    const float* W2 = (const float*)d_in[8];
    const float* b2 = (const float*)d_in[9];
    const float* g2 = (const float*)d_in[10];
    const float* be2 = (const float*)d_in[11];
    const float* m2 = (const float*)d_in[12];
    const float* v2 = (const float*)d_in[13];
    const float* W3 = (const float*)d_in[14];
    const float* b3 = (const float*)d_in[15];
    const float* g3 = (const float*)d_in[16];
    const float* be3 = (const float*)d_in[17];
    const float* m3 = (const float*)d_in[18];
    const float* v3 = (const float*)d_in[19];

    const int N = NPTS;
    const int B = in_sizes[0] / (3 * N);   // 8

    char* ws = (char*)d_ws;
    size_t off = 0;
    int* knn_idx = (int*)(ws + off);        off += (size_t)2 * B * N * KNN * sizeof(int);
    float* W1f   = (float*)(ws + off);      off += 64 * 4 * sizeof(float);
    float* bb1   = (float*)(ws + off);      off += 64 * sizeof(float);
    float* bb2   = (float*)(ws + off);      off += 64 * sizeof(float);
    float* bb3   = (float*)(ws + off);      off += 128 * sizeof(float);
    _Float16* W2A = (_Float16*)(ws + off);  off += 64 * 64 * sizeof(_Float16);
    _Float16* W3A = (_Float16*)(ws + off);  off += 128 * 64 * sizeof(_Float16);

    (void)hipFuncSetAttribute(reinterpret_cast<const void*>(&knn_kernel),
                              hipFuncAttributeMaxDynamicSharedMemorySize,
                              QT * N * (int)sizeof(float));

    init_weights_kernel<<<1, 256, 0, stream>>>(
        W1, b1, g1, be1, m1, v1, W2, b2, g2, be2, m2, v2,
        W3, b3, g3, be3, m3, v3, W1f, bb1, bb2, bb3, W2A, W3A);

    knn_kernel<<<dim3(N / QT, B, 2), 128, QT * N * sizeof(float), stream>>>(
        points1, points2, knn_idx, B, N);

    fuse_kernel<<<dim3(N, B), 32, 0, stream>>>(
        points1, points2, knn_idx, W1f, bb1, W2A, bb2, W3A, bb3,
        (float*)d_out, B, N);
}
